// LLaMA2Block_9148280340622
// MI455X (gfx1250) — matrix operand order repaired and checked by reading, not compiled, not run on hardware
//
#include <hip/hip_runtime.h>

typedef __attribute__((ext_vector_type(16))) __bf16 v16bf;
typedef __attribute__((ext_vector_type(8)))  float  v8f;

__device__ __forceinline__ __bf16 f2bf(float f) {
  union { float f; unsigned u; } in; in.f = f;
  unsigned r = in.u + 0x7FFFu + ((in.u >> 16) & 1u);   // round-to-nearest-even
  union { unsigned short s; __bf16 b; } out;
  out.s = (unsigned short)(r >> 16);
  return out.b;
}

// A-matrix (16x32 bf16) lane/element <-> K mapping (ISA 7.12.2):
// e 0..7  -> k = 8*half + e ; e 8..15 -> k = 16 + 8*half + (e-8)
__device__ __forceinline__ void kmapA(int k, int& half, int& e) {
  if      (k < 8)  { half = 0; e = k; }
  else if (k < 16) { half = 1; e = k - 8; }
  else if (k < 24) { half = 0; e = k - 8; }
  else             { half = 1; e = k - 16; }
}
// B-matrix (32x16 bf16): same k order as A (lanes 0-15: K 0..7,16..23; lanes 16-31: K 8..15,24..31)
// -> half = (k>>3)&1, e = (k&7) + ((k>>4)<<3)

// ---------------------------------------------------------------------------
// CDNA5 async global->LDS copy (GLOBAL_LOAD_ASYNC_TO_LDS_B128, ASYNCcnt).
// Per-lane global source and LDS destination. The LDS byte offset is the low
// 32 bits of the flat pointer (ISA 10.2, LDS aperture).
// ---------------------------------------------------------------------------
__device__ __forceinline__ void async_copy_b128(void* lds_dst, const void* gsrc) {
  const unsigned loff = (unsigned)(unsigned long long)lds_dst;
  asm volatile("global_load_async_to_lds_b128 %0, %1, off"
               :: "v"(loff), "v"((unsigned long long)gsrc)
               : "memory");
}

__device__ __forceinline__ void wait_async0() {
#if __has_builtin(__builtin_amdgcn_s_wait_asynccnt)
  __builtin_amdgcn_s_wait_asynccnt(0);
#else
  asm volatile("s_wait_asynccnt 0x0" ::: "memory");
#endif
}

// ---------------------------------------------------------------------------
// RMSNorm: y = w * x / (||x||/sqrt(H) + eps), output bf16
// ---------------------------------------------------------------------------
__global__ __launch_bounds__(256) void rmsnorm_kernel(
    const float* __restrict__ X, const float* __restrict__ w,
    __bf16* __restrict__ Y, int Hd) {
  const long row = blockIdx.x;
  const float* x = X + row * (long)Hd;
  float ss = 0.f;
  for (int i = threadIdx.x * 4; i < Hd; i += 1024) {
    float4 v = *(const float4*)(x + i);
    ss += v.x * v.x + v.y * v.y + v.z * v.z + v.w * v.w;
  }
  #pragma unroll
  for (int o = 16; o > 0; o >>= 1) ss += __shfl_xor(ss, o, 32);
  __shared__ float warps[8];
  if ((threadIdx.x & 31) == 0) warps[threadIdx.x >> 5] = ss;
  __syncthreads();
  float tot = 0.f;
  #pragma unroll
  for (int i = 0; i < 8; ++i) tot += warps[i];
  const float rms = sqrtf(tot / (float)Hd);
  const float inv = 1.0f / (rms + 1e-8f);
  __bf16* y = Y + row * (long)Hd;
  for (int i = threadIdx.x * 4; i < Hd; i += 1024) {
    float4 v = *(const float4*)(x + i);
    y[i + 0] = f2bf(v.x * w[i + 0] * inv);
    y[i + 1] = f2bf(v.y * w[i + 1] * inv);
    y[i + 2] = f2bf(v.z * w[i + 2] * inv);
    y[i + 3] = f2bf(v.w * w[i + 3] * inv);
  }
}

// ---------------------------------------------------------------------------
// GEMM: C[M,N] = A(bf16)[M,K] * W(fp32->bf16)[K,N] + bias (+res)
// Block tile 128x64, BK=32, 256 threads = 8 waves, wave -> 2x2 WMMA tiles.
// A tile staged via async global->LDS; W converted fp32->bf16 via DS stores.
// LDS holds tiles pre-swizzled into WMMA fragment order.
// ---------------------------------------------------------------------------
template<bool OUT_BF16, bool HAS_RES>
__global__ __launch_bounds__(256) void gemm_wmma(
    const __bf16* __restrict__ A, const float* __restrict__ W,
    const float* __restrict__ bias, const float* __restrict__ res,
    void* __restrict__ Cout, int M, int N, int K) {
  __shared__ __align__(16) __bf16 ldsA[8][32][16];
  __shared__ __align__(16) __bf16 ldsW[4][32][16];

  const int tid = threadIdx.x, lane = tid & 31, wave = tid >> 5;
  const int wm = wave >> 1, wn = wave & 1;
  const int bM = blockIdx.y * 128, bN = blockIdx.x * 64;

  v8f acc[2][2] = {};

  // A staging: thread -> (row, k-half of 16)
  const int arow = tid >> 1, akh = tid & 1;
  const int asub = arow >> 4, am = arow & 15;
  const long aoff = (long)(bM + arow) * K + akh * 16;
  // W staging: thread -> (k row, 8 cols)
  const int wk = tid >> 3, wc0 = (tid & 7) * 8;
  const int whalf = (wk >> 3) & 1, we = (wk & 7) + ((wk >> 4) << 3);  // B-matrix mapping
  const int wsub = wc0 >> 4, wlane0 = (wc0 & 15) + 16 * whalf;

  for (int kt = 0; kt < K; kt += 32) {
    __syncthreads();
    {  // stage A (already bf16) -> fragment order, async DMA to LDS
      const __bf16* p = A + aoff + kt;
      async_copy_b128(&ldsA[asub][am][akh * 8],      p);      // k 0..7 / 16..23
      async_copy_b128(&ldsA[asub][am + 16][akh * 8], p + 8);  // k 8..15 / 24..31
    }
    {  // stage W fp32 -> bf16, fragment order
      const float* p = W + (long)(kt + wk) * N + bN + wc0;
      float4 f0 = *(const float4*)p;
      float4 f1 = *(const float4*)(p + 4);
      __bf16 b8[8] = { f2bf(f0.x), f2bf(f0.y), f2bf(f0.z), f2bf(f0.w),
                       f2bf(f1.x), f2bf(f1.y), f2bf(f1.z), f2bf(f1.w) };
      #pragma unroll
      for (int j = 0; j < 8; ++j) ldsW[wsub][wlane0 + j][we] = b8[j];
      if (kt + 32 < K) __builtin_prefetch(p + (long)32 * N, 0, 1);
    }
    wait_async0();        // drain ASYNCcnt before signaling the barrier
    __syncthreads();
    v16bf af[2], bfr[2];
    af[0]  = *(const v16bf*)&ldsA[wm * 2 + 0][lane][0];
    af[1]  = *(const v16bf*)&ldsA[wm * 2 + 1][lane][0];
    bfr[0] = *(const v16bf*)&ldsW[wn * 2 + 0][lane][0];
    bfr[1] = *(const v16bf*)&ldsW[wn * 2 + 1][lane][0];
    #pragma unroll
    for (int i = 0; i < 2; ++i)
      #pragma unroll
      for (int j = 0; j < 2; ++j)
        acc[i][j] = __builtin_amdgcn_wmma_f32_16x16x32_bf16(
            false, af[i], false, bfr[j], (short)0, acc[i][j], false, false);
  }
  // epilogue (C layout: col = lane&15, row = vgpr + 8*(lane>>4))
  const int n_lo = lane & 15, hl = lane >> 4;
  #pragma unroll
  for (int i = 0; i < 2; ++i) {
    #pragma unroll
    for (int j = 0; j < 2; ++j) {
      const int col = bN + (wn * 2 + j) * 16 + n_lo;
      const float bv = bias[col];
      #pragma unroll
      for (int r = 0; r < 8; ++r) {
        const int row = bM + (wm * 2 + i) * 16 + hl * 8 + r;
        float v = acc[i][j][r] + bv;
        if (HAS_RES) v += res[(long)row * N + col];
        if (OUT_BF16) ((__bf16*)Cout)[(long)row * N + col] = f2bf(v);
        else          ((float*)Cout)[(long)row * N + col]  = v;
      }
    }
  }
}

// ---------------------------------------------------------------------------
// Fused gate/up GEMM: out = silu(A*Wg+bg) * (A*Wu+bu), bf16 out.
// ---------------------------------------------------------------------------
__global__ __launch_bounds__(256) void gemm_gateup(
    const __bf16* __restrict__ A,
    const float* __restrict__ Wg, const float* __restrict__ bg,
    const float* __restrict__ Wu, const float* __restrict__ bu,
    __bf16* __restrict__ Out, int M, int N, int K) {
  __shared__ __align__(16) __bf16 ldsA[8][32][16];
  __shared__ __align__(16) __bf16 ldsWg[4][32][16];
  __shared__ __align__(16) __bf16 ldsWu[4][32][16];

  const int tid = threadIdx.x, lane = tid & 31, wave = tid >> 5;
  const int wm = wave >> 1, wn = wave & 1;
  const int bM = blockIdx.y * 128, bN = blockIdx.x * 64;

  v8f accg[2][2] = {}, accu[2][2] = {};

  const int arow = tid >> 1, akh = tid & 1;
  const int asub = arow >> 4, am = arow & 15;
  const long aoff = (long)(bM + arow) * K + akh * 16;
  const int wk = tid >> 3, wc0 = (tid & 7) * 8;
  const int whalf = (wk >> 3) & 1, we = (wk & 7) + ((wk >> 4) << 3);
  const int wsub = wc0 >> 4, wlane0 = (wc0 & 15) + 16 * whalf;

  for (int kt = 0; kt < K; kt += 32) {
    __syncthreads();
    {
      const __bf16* p = A + aoff + kt;
      async_copy_b128(&ldsA[asub][am][akh * 8],      p);
      async_copy_b128(&ldsA[asub][am + 16][akh * 8], p + 8);
    }
    {
      const long wo = (long)(kt + wk) * N + bN + wc0;
      float4 g0 = *(const float4*)(Wg + wo), g1 = *(const float4*)(Wg + wo + 4);
      float4 u0 = *(const float4*)(Wu + wo), u1 = *(const float4*)(Wu + wo + 4);
      __bf16 gb[8] = { f2bf(g0.x), f2bf(g0.y), f2bf(g0.z), f2bf(g0.w),
                       f2bf(g1.x), f2bf(g1.y), f2bf(g1.z), f2bf(g1.w) };
      __bf16 ub[8] = { f2bf(u0.x), f2bf(u0.y), f2bf(u0.z), f2bf(u0.w),
                       f2bf(u1.x), f2bf(u1.y), f2bf(u1.z), f2bf(u1.w) };
      #pragma unroll
      for (int j = 0; j < 8; ++j) {
        ldsWg[wsub][wlane0 + j][we] = gb[j];
        ldsWu[wsub][wlane0 + j][we] = ub[j];
      }
    }
    wait_async0();
    __syncthreads();
    v16bf af[2], bg_[2], bu_[2];
    af[0]  = *(const v16bf*)&ldsA[wm * 2 + 0][lane][0];
    af[1]  = *(const v16bf*)&ldsA[wm * 2 + 1][lane][0];
    bg_[0] = *(const v16bf*)&ldsWg[wn * 2 + 0][lane][0];
    bg_[1] = *(const v16bf*)&ldsWg[wn * 2 + 1][lane][0];
    bu_[0] = *(const v16bf*)&ldsWu[wn * 2 + 0][lane][0];
    bu_[1] = *(const v16bf*)&ldsWu[wn * 2 + 1][lane][0];
    #pragma unroll
    for (int i = 0; i < 2; ++i)
      #pragma unroll
      for (int j = 0; j < 2; ++j) {
        accg[i][j] = __builtin_amdgcn_wmma_f32_16x16x32_bf16(
            false, af[i], false, bg_[j], (short)0, accg[i][j], false, false);
        accu[i][j] = __builtin_amdgcn_wmma_f32_16x16x32_bf16(
            false, af[i], false, bu_[j], (short)0, accu[i][j], false, false);
      }
  }
  const int n_lo = lane & 15, hl = lane >> 4;
  #pragma unroll
  for (int i = 0; i < 2; ++i) {
    #pragma unroll
    for (int j = 0; j < 2; ++j) {
      const int col = bN + (wn * 2 + j) * 16 + n_lo;
      const float bgv = bg[col], buv = bu[col];
      #pragma unroll
      for (int r = 0; r < 8; ++r) {
        const int row = bM + (wm * 2 + i) * 16 + hl * 8 + r;
        float g = accg[i][j][r] + bgv;
        float u = accu[i][j][r] + buv;
        float s = g / (1.0f + __expf(-g)) * u;   // silu(g) * u
        Out[(long)row * N + col] = f2bf(s);
      }
    }
  }
}

// ---------------------------------------------------------------------------
// Flash attention, hd = 128, causal. 128 threads = 4 waves, 16 q-rows/wave.
// Key tiles of 32; K staged via async global->LDS in B-fragment order;
// V staged via DS scatter; P bounced through LDS (C-layout -> A-layout).
// ---------------------------------------------------------------------------
__global__ __launch_bounds__(128) void flash_attn(
    const __bf16* __restrict__ Q, const __bf16* __restrict__ Kb,
    const __bf16* __restrict__ Vb, __bf16* __restrict__ O, int S, int NH) {
  constexpr int HD = 128;
  const int Hd = NH * HD;
  const int bh = blockIdx.y, b = bh / NH, h = bh % NH;
  const int qB = blockIdx.x * 64;
  const int tid = threadIdx.x, lane = tid & 31, wave = tid >> 5;
  const int qrow0 = qB + wave * 16;
  const long rowbase = (long)b * S;
  const int hoff = h * HD;

  __shared__ __align__(16) __bf16 ldsK[2][4][32][16];  // [keysub][dchunk]
  __shared__ __align__(16) __bf16 ldsV[8][32][16];     // [dsub]
  __shared__ __align__(16) __bf16 ldsP[4][32][16];     // per-wave P tile

  const int mloc = lane & 15, hl = lane >> 4;

  // Load Q A-fragments (4 x 16x32 over d)
  v16bf qf[4];
  {
    const __bf16* qrow = Q + (rowbase + qrow0 + mloc) * Hd + hoff;
    #pragma unroll
    for (int dc = 0; dc < 4; ++dc) {
      union { v16bf v; uint4 q[2]; } t;
      t.q[0] = *(const uint4*)(qrow + dc * 32 + 16 * hl);       // e0..7
      t.q[1] = *(const uint4*)(qrow + dc * 32 + 16 * hl + 8);   // e8..15
      qf[dc] = t.v;
    }
  }

  v8f of[8] = {};
  float mrow[8], lrow[8];
  #pragma unroll
  for (int r = 0; r < 8; ++r) { mrow[r] = -1e30f; lrow[r] = 0.f; }

  const int skey = tid >> 2, sq = tid & 3;             // staging split
  const int skc = skey & 15, sks = skey >> 4;
  const float scale = 0.08838834764831845f;            // 1/sqrt(128)
  const int nkt = (qB + 64) / 32;

  for (int kt = 0; kt < nkt; ++kt) {
    const int kbase = kt * 32;
    {  // stage K tile (scores-B: col=key, Kdim=d -> half=d>>4, e=d&15)
      const __bf16* kr = Kb + (rowbase + kbase + skey) * Hd + hoff + sq * 32;
      async_copy_b128(&ldsK[sks][sq][skc][0],      kr);        // d-local 0..7
      async_copy_b128(&ldsK[sks][sq][skc][8],      kr + 8);    // d-local 8..15
      async_copy_b128(&ldsK[sks][sq][skc + 16][0], kr + 16);   // d-local 16..23
      async_copy_b128(&ldsK[sks][sq][skc + 16][8], kr + 24);   // d-local 24..31
    }
    {  // stage V tile (AV-B: col=d, Kdim=key -> half=key>>4, e=key&15)
      const __bf16* vr = Vb + (rowbase + kbase + skey) * Hd + hoff + sq * 32;
      union { uint4 q[4]; __bf16 e[32]; } t;
      t.q[0] = *(const uint4*)(vr);
      t.q[1] = *(const uint4*)(vr + 8);
      t.q[2] = *(const uint4*)(vr + 16);
      t.q[3] = *(const uint4*)(vr + 24);
      #pragma unroll
      for (int jj = 0; jj < 32; ++jj) {
        const int d = sq * 32 + jj;
        ldsV[d >> 4][(d & 15) + 16 * sks][skc] = t.e[jj];
      }
    }
    wait_async0();
    __syncthreads();

    // scores: 16 q x 32 k as two 16x16 C tiles
    v8f s[2];
    #pragma unroll
    for (int ks = 0; ks < 2; ++ks) {
      v8f acc = {};
      #pragma unroll
      for (int dc = 0; dc < 4; ++dc)
        acc = __builtin_amdgcn_wmma_f32_16x16x32_bf16(
            false, qf[dc], false, *(const v16bf*)&ldsK[ks][dc][lane][0],
            (short)0, acc, false, false);
      s[ks] = acc;
    }
    // mask + scale + online softmax
    #pragma unroll
    for (int r = 0; r < 8; ++r) {
      const int qg = qrow0 + hl * 8 + r;
      #pragma unroll
      for (int ks = 0; ks < 2; ++ks) {
        const int kg = kbase + ks * 16 + mloc;
        const float v = s[ks][r] * scale;
        s[ks][r] = (kg <= qg) ? v : -1e30f;
      }
      float rm = fmaxf(s[0][r], s[1][r]);
      #pragma unroll
      for (int o = 1; o < 16; o <<= 1) rm = fmaxf(rm, __shfl_xor(rm, o, 32));
      const float mnew  = fmaxf(mrow[r], rm);
      const float alpha = __expf(mrow[r] - mnew);
      const float p0 = __expf(s[0][r] - mnew);
      const float p1 = __expf(s[1][r] - mnew);
      s[0][r] = p0; s[1][r] = p1;
      float rs = p0 + p1;
      #pragma unroll
      for (int o = 1; o < 16; o <<= 1) rs += __shfl_xor(rs, o, 32);
      lrow[r] = lrow[r] * alpha + rs;
      mrow[r] = mnew;
      #pragma unroll
      for (int ds = 0; ds < 8; ++ds) of[ds][r] = of[ds][r] * alpha;
    }
    // P: C-layout -> A-fragment layout via LDS bounce
    #pragma unroll
    for (int ks = 0; ks < 2; ++ks) {
      int ph = ks, pe = mloc;
      #pragma unroll
      for (int r = 0; r < 8; ++r)
        ldsP[wave][(hl * 8 + r) + 16 * ph][pe] = f2bf(s[ks][r]);
    }
    __syncthreads();
    const v16bf pf = *(const v16bf*)&ldsP[wave][lane][0];
    #pragma unroll
    for (int ds = 0; ds < 8; ++ds)
      of[ds] = __builtin_amdgcn_wmma_f32_16x16x32_bf16(
          false, pf, false, *(const v16bf*)&ldsV[ds][lane][0],
          (short)0, of[ds], false, false);
    __syncthreads();
  }
  // normalize + store bf16
  #pragma unroll
  for (int r = 0; r < 8; ++r) {
    const float inv = 1.0f / lrow[r];
    const long orow = (rowbase + qrow0 + hl * 8 + r) * Hd + hoff;
    #pragma unroll
    for (int ds = 0; ds < 8; ++ds)
      O[orow + ds * 16 + mloc] = f2bf(of[ds][r] * inv);
  }
}

// ---------------------------------------------------------------------------
extern "C" void kernel_launch(void* const* d_in, const int* in_sizes, int n_in,
                              void* d_out, int out_size, void* d_ws, size_t ws_size,
                              hipStream_t stream) {
  constexpr int B = 2, S = 2048, H = 2048, NH = 16, F = 8192;
  constexpr long M = (long)B * S;

  const float* x           = (const float*)d_in[0];
  /* d_in[1] = attn_mask (causal tril) -> implemented analytically */
  const float* wq          = (const float*)d_in[2];
  const float* bq          = (const float*)d_in[3];
  const float* wk          = (const float*)d_in[4];
  const float* bk          = (const float*)d_in[5];
  const float* wv          = (const float*)d_in[6];
  const float* bv          = (const float*)d_in[7];
  const float* wo          = (const float*)d_in[8];
  const float* bo          = (const float*)d_in[9];
  const float* attn_norm_w = (const float*)d_in[10];
  const float* ffn_norm_w  = (const float*)d_in[11];
  const float* w_gate      = (const float*)d_in[12];
  const float* b_gate      = (const float*)d_in[13];
  const float* w_up        = (const float*)d_in[14];
  const float* b_up        = (const float*)d_in[15];
  const float* w_down      = (const float*)d_in[16];
  const float* b_down      = (const float*)d_in[17];

  char* ws = (char*)d_ws;
  size_t off = 0;
  auto take = [&](size_t bytes) -> void* {
    void* p = ws + off;
    off = (off + bytes + 255) & ~(size_t)255;
    return p;
  };
  __bf16* xn = (__bf16*)take(M * H * 2);
  __bf16* qb = (__bf16*)take(M * H * 2);
  __bf16* kb = (__bf16*)take(M * H * 2);
  __bf16* vb = (__bf16*)take(M * H * 2);
  __bf16* ao = (__bf16*)take(M * H * 2);
  float*  hb = (float*)take(M * H * 4);
  __bf16* hn = (__bf16*)take(M * H * 2);
  __bf16* gu = (__bf16*)take(M * F * 2);

  const dim3 gH(H / 64, M / 128);
  const dim3 gF(F / 64, M / 128);

  // 1) pre-attn RMSNorm -> bf16
  rmsnorm_kernel<<<(unsigned)M, 256, 0, stream>>>(x, attn_norm_w, xn, H);
  // 2) Q, K, V projections (bf16 out)
  gemm_wmma<true,  false><<<gH, 256, 0, stream>>>(xn, wq, bq, nullptr, qb, (int)M, H, H);
  gemm_wmma<true,  false><<<gH, 256, 0, stream>>>(xn, wk, bk, nullptr, kb, (int)M, H, H);
  gemm_wmma<true,  false><<<gH, 256, 0, stream>>>(xn, wv, bv, nullptr, vb, (int)M, H, H);
  // 3) causal flash attention
  flash_attn<<<dim3(S / 64, B * NH), 128, 0, stream>>>(qb, kb, vb, ao, S, NH);
  // 4) output projection + residual -> h (fp32)
  gemm_wmma<false, true><<<gH, 256, 0, stream>>>(ao, wo, bo, x, hb, (int)M, H, H);
  // 5) pre-FFN RMSNorm -> bf16
  rmsnorm_kernel<<<(unsigned)M, 256, 0, stream>>>(hb, ffn_norm_w, hn, H);
  // 6) fused gate/up + SiLU*mul -> bf16
  gemm_gateup<<<gF, 256, 0, stream>>>(hn, w_gate, b_gate, w_up, b_up, gu, (int)M, F, H);
  // 7) down projection + residual -> d_out (fp32)
  gemm_wmma<false, true><<<gH, 256, 0, stream>>>(gu, w_down, b_down, hb,
                                                 (float*)d_out, (int)M, H, F);
}
